// Decoder_LSTM_36438502540157
// MI455X (gfx1250) — matrix operand order repaired and checked by reading, not compiled, not run on hardware
//
#include <hip/hip_runtime.h>
#include <hip/hip_bf16.h>

// ---------------------------------------------------------------------------
// Sizes (fixed by the reference)
// ---------------------------------------------------------------------------
#define B_    64
#define P_    196
#define ENC_  2048
#define DEC_  512
#define ATT_  512
#define E_    300
#define V_    30000
#define T_    21
#define STEPS 20
#define KCAT  2880   // E_+ENC_+DEC_ = 2860 padded to a multiple of 32
#define NSUB  4      // N-subtiles (16 cols each) per wave

typedef __attribute__((ext_vector_type(16))) __bf16 v16bf;
typedef __attribute__((ext_vector_type(8)))  float  v8f;

static inline int ntiles_pad(int N) { return ((N / 16 + NSUB - 1) / NSUB) * NSUB; }

// ---------------------------------------------------------------------------
// WMMA GEMM with pre-packed B:  C[M,N] = A[M,K](fp32) * B(bf16, packed) + bias
//
// B is stored as 32x16 (KxN) tiles in a contiguous-k (not AMD's table order) v_wmma_f32_16x16x32_bf16
// source-fragment layout: tile index = ntile*(K/32)+ktile; within a tile,
// lane l's 16 elements are contiguous (32 B) at offset l*16:
//   element e of lane l -> B[kt*32 + (l>>4)*16 + e][nt*16 + (l&15)]
// The ntile axis is zero-padded to a multiple of NSUB so the k-loop has no
// guards; only the C store checks nc < N.
//
// A (fp32) is fetched per lane as one contiguous 16-float run (k = 16*half + 0..15) and converted
// on the fly (v_cvt_pk_bf16_f32 co-executes with the XDL WMMA pipeline).
// Each wave computes NSUB 16x16 output tiles, reusing one A fragment for
// NSUB back-to-back WMMAs: 12 b128 loads per 4 matrix ops.
// No LDS, no barriers.
// ---------------------------------------------------------------------------
__global__ __launch_bounds__(256)
void gemm_bf16_wmma(const float* __restrict__ A, int lda,
                    const __bf16* __restrict__ Bp,
                    const float* __restrict__ bias,
                    float* __restrict__ C, long long ldc,
                    int M, int N, int K,
                    const int* __restrict__ rowmap)
{
    const int lane = threadIdx.x & 31;
    const int wv   = threadIdx.x >> 5;
    const int m0   = blockIdx.x * 16;
    const int ntp  = ((N / 16 + NSUB - 1) / NSUB) * NSUB;
    const int nt0  = (blockIdx.y * 8 + wv) * NSUB;
    if (nt0 >= ntp) return;                      // wave-uniform exit
    const int ktiles = K >> 5;

    int arow = m0 + (lane & 15);
    if (rowmap) arow = rowmap[arow];
    const int half = lane >> 4;
    const float*  Abase = A + (size_t)arow * lda + half * 16;
    const __bf16* Bt    = Bp + (size_t)nt0 * ktiles * 512 + lane * 16;
    const size_t  bstr  = (size_t)ktiles * 512;  // elements between ntile streams

    v8f acc[NSUB];
#pragma unroll
    for (int j = 0; j < NSUB; ++j)
        acc[j] = (v8f){0.f, 0.f, 0.f, 0.f, 0.f, 0.f, 0.f, 0.f};

#pragma unroll 2
    for (int kt = 0; kt < ktiles; ++kt) {
        const float* a0 = Abase + kt * 32;
        const float4 r00 = *reinterpret_cast<const float4*>(a0);
        const float4 r01 = *reinterpret_cast<const float4*>(a0 + 4);
        const float4 r10 = *reinterpret_cast<const float4*>(a0 + 8);
        const float4 r11 = *reinterpret_cast<const float4*>(a0 + 12);

        v16bf bfr[NSUB];
#pragma unroll
        for (int j = 0; j < NSUB; ++j)
            bfr[j] = *reinterpret_cast<const v16bf*>(Bt + j * bstr + (size_t)kt * 512);
        if (kt + 2 < ktiles) {
#pragma unroll
            for (int j = 0; j < NSUB; ++j)
                __builtin_prefetch((const void*)(Bt + j * bstr + (size_t)(kt + 2) * 512), 0, 1);
        }

        v16bf af;
        af[0]  = (__bf16)r00.x; af[1]  = (__bf16)r00.y;
        af[2]  = (__bf16)r00.z; af[3]  = (__bf16)r00.w;
        af[4]  = (__bf16)r01.x; af[5]  = (__bf16)r01.y;
        af[6]  = (__bf16)r01.z; af[7]  = (__bf16)r01.w;
        af[8]  = (__bf16)r10.x; af[9]  = (__bf16)r10.y;
        af[10] = (__bf16)r10.z; af[11] = (__bf16)r10.w;
        af[12] = (__bf16)r11.x; af[13] = (__bf16)r11.y;
        af[14] = (__bf16)r11.z; af[15] = (__bf16)r11.w;

#pragma unroll
        for (int j = 0; j < NSUB; ++j)
            acc[j] = __builtin_amdgcn_wmma_f32_16x16x32_bf16(
                         false, af, false, bfr[j], (short)0, acc[j], false, false);
    }

    // store: VGPR r -> M = r + 8*(lane>>4), N = lane&15 (within each subtile)
    const int mb = (lane >> 4) * 8;
#pragma unroll
    for (int j = 0; j < NSUB; ++j) {
        const int nc = (nt0 + j) * 16 + (lane & 15);
        if (nc < N) {
            const float bv = bias ? bias[nc] : 0.f;
#pragma unroll
            for (int r = 0; r < 8; ++r)
                C[(size_t)(m0 + mb + r) * ldc + nc] = acc[j][r] + bv;
        }
    }
}

// ---------------------------------------------------------------------------
// Pack fp32 [K,N] row-major weights into bf16 WMMA B-fragment tile layout,
// zero-padding the ntile axis to a multiple of NSUB.
// ---------------------------------------------------------------------------
__global__ void packB_kernel(const float* __restrict__ W, __bf16* __restrict__ out,
                             int K, int N, int ntp)
{
    const long long total = (long long)(K >> 5) * ntp * 512;
    const long long i = (long long)blockIdx.x * blockDim.x + threadIdx.x;
    if (i >= total) return;
    const int       r    = (int)(i & 511);
    const long long tile = i >> 9;
    const int ktiles = K >> 5;
    const int nt = (int)(tile / ktiles);
    const int kt = (int)(tile % ktiles);
    const int lane = r >> 4, e = r & 15;
    const int k = kt * 32 + (lane >> 4) * 16 + e;
    const int n = nt * 16 + (lane & 15);
    out[i] = (n < N) ? (__bf16)W[(size_t)k * N + n] : (__bf16)0.f;
}

// Same packing for the fused-gates B matrix built from W_ih^T / W_hh^T / pad:
// logical B[k][n]: k<2348 -> W_ih[n][k]; k<2860 -> W_hh[n][k-2348]; else 0.
__global__ void packWcat_kernel(const float* __restrict__ W_ih,
                                const float* __restrict__ W_hh,
                                __bf16* __restrict__ out)
{
    const long long i = (long long)blockIdx.x * blockDim.x + threadIdx.x;
    if (i >= (long long)KCAT * (4 * DEC_)) return;
    const int       r    = (int)(i & 511);
    const long long tile = i >> 9;
    const int ktiles = KCAT >> 5;
    const int nt = (int)(tile / ktiles);
    const int kt = (int)(tile % ktiles);
    const int lane = r >> 4, e = r & 15;
    const int k = kt * 32 + (lane >> 4) * 16 + e;
    const int n = nt * 16 + (lane & 15);
    float v = 0.f;
    if (k < E_ + ENC_)              v = W_ih[(size_t)n * (E_ + ENC_) + k];
    else if (k < E_ + ENC_ + DEC_)  v = W_hh[(size_t)n * DEC_ + (k - (E_ + ENC_))];
    out[i] = (__bf16)v;
}

// ---------------------------------------------------------------------------
// Stable descending argsort of caption_len (B=64, one block), plus the
// permuted captions / seq_length outputs (as float, per output layout).
// ---------------------------------------------------------------------------
__global__ void sort_kernel(const int* __restrict__ captions,
                            const int* __restrict__ caption_len,
                            int* __restrict__ order, int* __restrict__ cap_s,
                            float* __restrict__ out_cap, float* __restrict__ out_seq)
{
    const int i = threadIdx.x;              // 64 threads
    __shared__ int ord[B_];
    const int li = caption_len[i];
    int rank = 0;
    for (int j = 0; j < B_; ++j) {
        const int lj = caption_len[j];
        if (lj > li || (lj == li && j < i)) rank++;
    }
    ord[rank] = i;
    __syncthreads();
    const int src = ord[i];
    order[i] = src;
    for (int j = 0; j < T_; ++j) {
        const int cv = captions[src * T_ + j];
        cap_s[i * T_ + j]   = cv;
        out_cap[i * T_ + j] = (float)cv;
    }
    out_seq[i] = (float)(caption_len[src] - 1);
}

__global__ void rowmap_kernel(const int* __restrict__ order, int* __restrict__ rowmap)
{
    const int i = blockIdx.x * blockDim.x + threadIdx.x;
    if (i < B_ * P_) rowmap[i] = order[i / P_] * P_ + (i % P_);
}

__global__ void mean_kernel(const float* __restrict__ features,
                            const int* __restrict__ order,
                            float* __restrict__ mean_f)
{
    const int b = blockIdx.x;
    const float* fb = features + (size_t)order[b] * P_ * ENC_;
    for (int d = threadIdx.x; d < ENC_; d += 256) {
        float s = 0.f;
        for (int p = 0; p < P_; ++p) s += fb[(size_t)p * ENC_ + d];
        mean_f[(size_t)b * ENC_ + d] = s * (1.f / (float)P_);
    }
}

__global__ void biasadd_kernel(const float* __restrict__ a, const float* __restrict__ b,
                               float* __restrict__ out, int n)
{
    const int i = blockIdx.x * blockDim.x + threadIdx.x;
    if (i < n) out[i] = a[i] + b[i];
}

// ---------------------------------------------------------------------------
// Additive attention: one block per batch row.
// ---------------------------------------------------------------------------
__global__ __launch_bounds__(256)
void attention_kernel(const float* __restrict__ enc_proj,
                      const float* __restrict__ dec_proj,
                      const float* __restrict__ W_att,
                      const float* __restrict__ b_att,
                      const float* __restrict__ features,
                      const int* __restrict__ order,
                      float* __restrict__ ctx,
                      float* __restrict__ alphas_out, int t)
{
    const int b   = blockIdx.x;
    const int tid = threadIdx.x;
    const int lane = tid & 31, wv = tid >> 5;

    __shared__ float dec[ATT_];
    __shared__ float sc[256];
    __shared__ float red[256];

    for (int d = tid; d < ATT_; d += 256) dec[d] = dec_proj[(size_t)b * ATT_ + d];
    __syncthreads();

    const float* ep = enc_proj + (size_t)b * P_ * ATT_;
    for (int p = wv; p < P_; p += 8) {
        float s = 0.f;
        const float* row = ep + (size_t)p * ATT_;
        for (int d = lane; d < ATT_; d += 32)
            s += fmaxf(row[d] + dec[d], 0.f) * W_att[d];
        for (int o = 16; o > 0; o >>= 1) s += __shfl_down(s, o, 32);
        if (lane == 0) sc[p] = s + b_att[0];
    }
    __syncthreads();

    // softmax over 196 positions
    float v = (tid < P_) ? sc[tid] : -3.4e38f;
    red[tid] = v; __syncthreads();
    for (int o = 128; o > 0; o >>= 1) {
        if (tid < o) red[tid] = fmaxf(red[tid], red[tid + o]);
        __syncthreads();
    }
    const float mx = red[0]; __syncthreads();
    const float ex = (tid < P_) ? __expf(v - mx) : 0.f;
    red[tid] = ex; __syncthreads();
    for (int o = 128; o > 0; o >>= 1) {
        if (tid < o) red[tid] += red[tid + o];
        __syncthreads();
    }
    const float inv = 1.f / red[0];
    __syncthreads();
    if (tid < P_) {
        const float a = ex * inv;
        sc[tid] = a;
        alphas_out[(size_t)b * STEPS * P_ + (size_t)t * P_ + tid] = a;
    }
    __syncthreads();

    // context vector
    const float* fb = features + (size_t)order[b] * P_ * ENC_;
    for (int d = tid; d < ENC_; d += 256) {
        float acc = 0.f;
        for (int p = 0; p < P_; ++p) acc += sc[p] * fb[(size_t)p * ENC_ + d];
        ctx[(size_t)b * ENC_ + d] = acc;
    }
}

// x = [embed(token_t) | ctx | h_prev | 0-pad]  (fp32, K = 2880)
__global__ void xcat_kernel(const float* __restrict__ emb, const int* __restrict__ cap_s,
                            const float* __restrict__ ctx, const float* __restrict__ h,
                            float* __restrict__ xcat, int t)
{
    const int i = blockIdx.x * blockDim.x + threadIdx.x;
    if (i >= B_ * KCAT) return;
    const int b = i / KCAT, k = i % KCAT;
    float v;
    if (k < E_)                    v = emb[(size_t)cap_s[b * T_ + t] * E_ + k];
    else if (k < E_ + ENC_)        v = ctx[(size_t)b * ENC_ + (k - E_)];
    else if (k < E_ + ENC_ + DEC_) v = h[(size_t)b * DEC_ + (k - (E_ + ENC_))];
    else                           v = 0.f;
    xcat[i] = v;
}

__global__ void lstm_kernel(const float* __restrict__ gates,
                            float* __restrict__ c, float* __restrict__ h)
{
    const int i = blockIdx.x * blockDim.x + threadIdx.x;
    if (i >= B_ * DEC_) return;
    const int b = i / DEC_, j = i % DEC_;
    const float* g = gates + (size_t)b * 4 * DEC_;
    const float ig = 1.f / (1.f + __expf(-g[j]));
    const float fg = 1.f / (1.f + __expf(-g[DEC_ + j]));
    const float gg = tanhf(g[2 * DEC_ + j]);
    const float og = 1.f / (1.f + __expf(-g[3 * DEC_ + j]));
    const float c2 = fg * c[i] + ig * gg;
    c[i] = c2;
    h[i] = og * tanhf(c2);
}

// ---------------------------------------------------------------------------
extern "C" void kernel_launch(void* const* d_in, const int* in_sizes, int n_in,
                              void* d_out, int out_size, void* d_ws, size_t ws_size,
                              hipStream_t stream)
{
    (void)in_sizes; (void)n_in; (void)out_size; (void)ws_size;

    const float* features    = (const float*)d_in[0];
    const int*   captions    = (const int*)d_in[1];
    const int*   caption_len = (const int*)d_in[2];
    const float* emb   = (const float*)d_in[3];
    const float* W_enc = (const float*)d_in[4];
    const float* b_enc = (const float*)d_in[5];
    const float* W_dec = (const float*)d_in[6];
    const float* b_dec = (const float*)d_in[7];
    const float* W_att = (const float*)d_in[8];
    const float* b_att = (const float*)d_in[9];
    const float* W_h0  = (const float*)d_in[10];
    const float* b_h0  = (const float*)d_in[11];
    const float* W_c0  = (const float*)d_in[12];
    const float* b_c0  = (const float*)d_in[13];
    const float* W_ih  = (const float*)d_in[14];
    const float* b_ih  = (const float*)d_in[15];
    const float* W_hh  = (const float*)d_in[16];
    const float* b_hh  = (const float*)d_in[17];
    const float* W_out = (const float*)d_in[18];
    const float* b_out = (const float*)d_in[19];

    float* out        = (float*)d_out;
    float* out_logits = out;                                        // [B][STEPS][V]
    float* out_alphas = out + (size_t)B_ * STEPS * V_;              // [B][STEPS][P]
    float* out_caps   = out_alphas + (size_t)B_ * STEPS * P_;       // [B][T]
    float* out_seq    = out_caps + (size_t)B_ * T_;                 // [B]

    char* ws = (char*)d_ws;
    size_t off = 0;
    auto alloc = [&](size_t bytes) -> char* {
        char* p = ws + off;
        off = (off + bytes + 255) & ~(size_t)255;
        return p;
    };
    const int ntpATT = ntiles_pad(ATT_);        // 32
    const int ntpDEC = ntiles_pad(DEC_);        // 32
    const int ntpG   = ntiles_pad(4 * DEC_);    // 128
    const int ntpV   = ntiles_pad(V_);          // 1876

    int*    order    = (int*)   alloc(B_ * 4);
    int*    cap_s    = (int*)   alloc((size_t)B_ * T_ * 4);
    int*    rowmap   = (int*)   alloc((size_t)B_ * P_ * 4);
    float*  mean_f   = (float*) alloc((size_t)B_ * ENC_ * 4);
    float*  h        = (float*) alloc((size_t)B_ * DEC_ * 4);
    float*  c        = (float*) alloc((size_t)B_ * DEC_ * 4);
    float*  dec_proj = (float*) alloc((size_t)B_ * ATT_ * 4);
    float*  ctx      = (float*) alloc((size_t)B_ * ENC_ * 4);
    float*  xcat     = (float*) alloc((size_t)B_ * KCAT * 4);
    float*  gates    = (float*) alloc((size_t)B_ * 4 * DEC_ * 4);
    float*  enc_proj = (float*) alloc((size_t)B_ * P_ * ATT_ * 4);
    float*  bias_g   = (float*) alloc((size_t)4 * DEC_ * 4);
    __bf16* Wenc_bf  = (__bf16*)alloc((size_t)ENC_ * ntpATT * 16 * 2);
    __bf16* Wdec_bf  = (__bf16*)alloc((size_t)DEC_ * ntpATT * 16 * 2);
    __bf16* Wh0_bf   = (__bf16*)alloc((size_t)ENC_ * ntpDEC * 16 * 2);
    __bf16* Wc0_bf   = (__bf16*)alloc((size_t)ENC_ * ntpDEC * 16 * 2);
    __bf16* Wcat_bf  = (__bf16*)alloc((size_t)KCAT * ntpG * 16 * 2);
    __bf16* Wout_bf  = (__bf16*)alloc((size_t)DEC_ * ntpV * 16 * 2);

    // --- one-time prep ---
    sort_kernel<<<1, B_, 0, stream>>>(captions, caption_len, order, cap_s, out_caps, out_seq);
    rowmap_kernel<<<(B_ * P_ + 255) / 256, 256, 0, stream>>>(order, rowmap);
    mean_kernel<<<B_, 256, 0, stream>>>(features, order, mean_f);

    auto packB = [&](const float* src, __bf16* dst, int K, int N) {
        const long long n = (long long)(K >> 5) * ntiles_pad(N) * 512;
        packB_kernel<<<(unsigned)((n + 255) / 256), 256, 0, stream>>>(src, dst, K, N, ntiles_pad(N));
    };
    packB(W_enc, Wenc_bf, ENC_, ATT_);
    packB(W_dec, Wdec_bf, DEC_, ATT_);
    packB(W_h0,  Wh0_bf,  ENC_, DEC_);
    packB(W_c0,  Wc0_bf,  ENC_, DEC_);
    packB(W_out, Wout_bf, DEC_, V_);
    {
        const long long n = (long long)KCAT * 4 * DEC_;   // ntpG == 4*DEC_/16, no pad
        packWcat_kernel<<<(unsigned)((n + 255) / 256), 256, 0, stream>>>(W_ih, W_hh, Wcat_bf);
    }
    biasadd_kernel<<<(4 * DEC_ + 255) / 256, 256, 0, stream>>>(b_ih, b_hh, bias_g, 4 * DEC_);

    auto launch_gemm = [&](const float* A, int lda, const __bf16* Bp, const float* bias,
                           float* C, long long ldc, int M, int N, int K, const int* rmap) {
        const int ntp = ntiles_pad(N);
        dim3 g(M / 16, (ntp / NSUB + 7) / 8);
        gemm_bf16_wmma<<<g, 256, 0, stream>>>(A, lda, Bp, bias, C, ldc, M, N, K, rmap);
    };

    // enc_proj = features[order] @ W_enc + b_enc   (M=12544, N=512, K=2048)
    launch_gemm(features, ENC_, Wenc_bf, b_enc, enc_proj, ATT_, B_ * P_, ATT_, ENC_, rowmap);
    // h0 / c0
    launch_gemm(mean_f, ENC_, Wh0_bf, b_h0, h, DEC_, B_, DEC_, ENC_, nullptr);
    launch_gemm(mean_f, ENC_, Wc0_bf, b_c0, c, DEC_, B_, DEC_, ENC_, nullptr);

    // --- 20 serial decode steps ---
    for (int t = 0; t < STEPS; ++t) {
        // dec_proj = h @ W_dec + b_dec
        launch_gemm(h, DEC_, Wdec_bf, b_dec, dec_proj, ATT_, B_, ATT_, DEC_, nullptr);
        attention_kernel<<<B_, 256, 0, stream>>>(enc_proj, dec_proj, W_att, b_att,
                                                 features, order, ctx, out_alphas, t);
        xcat_kernel<<<(B_ * KCAT + 255) / 256, 256, 0, stream>>>(emb, cap_s, ctx, h, xcat, t);
        // gates = [x|h] @ [W_ih^T; W_hh^T] + (b_ih + b_hh)
        launch_gemm(xcat, KCAT, Wcat_bf, bias_g, gates, 4 * DEC_, B_, 4 * DEC_, KCAT, nullptr);
        lstm_kernel<<<(B_ * DEC_ + 255) / 256, 256, 0, stream>>>(gates, c, h);
        // logits -> d_out[t] directly; row stride = STEPS*V
        launch_gemm(h, DEC_, Wout_bf, b_out, out_logits + (size_t)t * V_, (long long)STEPS * V_,
                    B_, V_, DEC_, nullptr);
    }
}
